// NerfBlock_56573309224351
// MI455X (gfx1250) — matrix operand order repaired and checked by reading, not compiled, not run on hardware
//
#include <hip/hip_runtime.h>
#include <hip/hip_bf16.h>
#include <math.h>

// ---------------------------------------------------------------------------
// Problem constants (reference: B=32, N=4096, HX=128, HS=1024, R=4)
// ---------------------------------------------------------------------------
#define B_      32
#define N_      4096
#define HX_     128
#define HS_     1024
#define HXR_    512                 // HX*R
#define PCOLS_  (2 * HX_ * HXR_)    // 131072 hypernet output columns
#define FCHALF_ (HX_ * HXR_)        // 65536 per-sample elements per fc

typedef __bf16 bf16_t;
typedef __attribute__((ext_vector_type(16))) __bf16 v16bf;
typedef __attribute__((ext_vector_type(8)))  __bf16 v8bf;
typedef __attribute__((ext_vector_type(4)))  __bf16 v4bf;
typedef __attribute__((ext_vector_type(8)))  float  v8f;
typedef __attribute__((ext_vector_type(4)))  float  v4f;

// ---------------------------------------------------------------------------
// WMMA helpers (CDNA5 wave32, D = A(16x32 bf16) * B(32x16 bf16) + C(16x16 f32))
// ---------------------------------------------------------------------------
static __device__ inline v16bf make_v16(v8bf lo, v8bf hi) {
  v16bf r;
#pragma unroll
  for (int i = 0; i < 8; ++i) { r[i] = lo[i]; r[8 + i] = hi[i]; }
  return r;
}

// A fragment: lane (l&15) holds row; 16 contiguous K: 0..15 for l<16,
// 16..31 for l>=16 (matches B fragment K order).
static __device__ inline v16bf load_a16x32(const bf16_t* base, int rowStride,
                                           int row0, int k0, int lane) {
  const bf16_t* p = base + (size_t)(row0 + (lane & 15)) * rowStride
                        + k0 + ((lane < 16) ? 0 : 16);
  v8bf lo = *(const v8bf*)p;
  v8bf hi = *(const v8bf*)(p + 8);
  return make_v16(lo, hi);
}

// B fragment: lane (l&15) = column; 16 contiguous K starting at 0 (l<16) or 16.
// 'p' must already point at this lane's 16-element run (16B-aligned).
static __device__ inline v16bf load_b32x16(const bf16_t* p) {
  v8bf lo = *(const v8bf*)p;
  v8bf hi = *(const v8bf*)(p + 8);
  return make_v16(lo, hi);
}

static __device__ inline v8f wmma_bf16(v16bf a, v16bf b, v8f c) {
  return __builtin_amdgcn_wmma_f32_16x16x32_bf16(false, a, false, b,
                                                 (short)0, c, false, false);
}

// ---------------------------------------------------------------------------
// Kernel 1: hypernetwork GEMM  params[32,131072] = s @ W + bias
// Streams W once (537 MB -> HBM-roofline, ~23 us @ 23.3 TB/s). W loads are
// non-temporal (single-use stream; keep L2 for the reused fc weights).
// Double-buffered LDS staging overlaps the next tile's global loads with the
// current tile's WMMAs. Each staging thread owns a 4(k)x4(j) sub-tile so the
// bf16 transpose packs into 4x ds_store_b64 (vs 16x ds_store_b16).
// Results stored pre-transposed:
//   fc1t[b][f][d]  (so fc1 B-fragments over K=d are contiguous)
//   fc2t[b][d][f]  (so fc2 B-fragments over K=f are contiguous)
// ---------------------------------------------------------------------------
__global__ __launch_bounds__(256) void hyper_gemm(const float* __restrict__ s,
                                                  const float* __restrict__ W,
                                                  const float* __restrict__ bias,
                                                  bf16_t* __restrict__ fc1t,
                                                  bf16_t* __restrict__ fc2t) {
  __shared__ bf16_t wt[2][128 * 32];   // double-buffered W tile: [j_local][k_local]

  const int tid  = threadIdx.x;
  const int wave = tid >> 5;
  const int lane = tid & 31;
  const int j0   = blockIdx.x * 128;
  const int lcol = lane & 15;
  const int khA  = (lane < 16) ? 0 : 16;   // A-fragment K sub-offset
  const int khB  = (lane < 16) ? 0 : 16;   // B-fragment K sub-offset

  v8f acc0 = {};   // samples 0..15
  v8f acc1 = {};   // samples 16..31

  // Staging: thread owns k rows [4*rgrp, 4*rgrp+4) x j cols [jst, jst+4)
  const int rgrp = tid >> 5;          // 0..7  -> k base = 4*rgrp
  const int jst  = (tid & 31) * 4;    // 0..124

  v4f reg[4];                          // reg[rr] = row (4*rgrp+rr), 4 j values

#define LOAD_TILE(k0)                                                        \
  {                                                                          \
    const float* wbase = W + (size_t)((k0) + 4 * rgrp) * PCOLS_ + j0 + jst;  \
    __builtin_prefetch(wbase + (size_t)32 * PCOLS_, 0, 1);                   \
    _Pragma("unroll")                                                        \
    for (int rr = 0; rr < 4; ++rr)                                           \
      reg[rr] = __builtin_nontemporal_load(                                  \
          (const v4f*)(wbase + (size_t)rr * PCOLS_));                        \
  }

#define STORE_TILE(buf)                                                      \
  {                                                                          \
    _Pragma("unroll")                                                        \
    for (int j = 0; j < 4; ++j) {                                            \
      v4bf pk;                                                               \
      pk[0] = (bf16_t)reg[0][j]; pk[1] = (bf16_t)reg[1][j];                  \
      pk[2] = (bf16_t)reg[2][j]; pk[3] = (bf16_t)reg[3][j];                  \
      *(v4bf*)&wt[buf][(jst + j) * 32 + 4 * rgrp] = pk;                      \
    }                                                                        \
  }

  LOAD_TILE(0);
  STORE_TILE(0);
  __syncthreads();

  for (int k0 = 0; k0 < HS_; k0 += 32) {
    const int buf   = (k0 >> 5) & 1;
    const bool more = (k0 + 32 < HS_);
    if (more) LOAD_TILE(k0 + 32);       // overlap next stream with compute

    // A fragments from s (tiny, cache resident), fp32 -> bf16 in-register
    const float* sp0 = s + (size_t)lcol * HS_ + k0 + khA;
    const float* sp1 = sp0 + (size_t)16 * HS_;
    v16bf a0, a1;
    {
      v4f f0 = *(const v4f*)(sp0);      v4f f1 = *(const v4f*)(sp0 + 4);
      v4f f2 = *(const v4f*)(sp0 + 8); v4f f3 = *(const v4f*)(sp0 + 12);
      v4f g0 = *(const v4f*)(sp1);      v4f g1 = *(const v4f*)(sp1 + 4);
      v4f g2 = *(const v4f*)(sp1 + 8); v4f g3 = *(const v4f*)(sp1 + 12);
#pragma unroll
      for (int e = 0; e < 4; ++e) {
        a0[e]      = (bf16_t)f0[e]; a0[4 + e]  = (bf16_t)f1[e];
        a0[8 + e]  = (bf16_t)f2[e]; a0[12 + e] = (bf16_t)f3[e];
        a1[e]      = (bf16_t)g0[e]; a1[4 + e]  = (bf16_t)g1[e];
        a1[8 + e]  = (bf16_t)g2[e]; a1[12 + e] = (bf16_t)g3[e];
      }
    }
    v16bf bb = load_b32x16(&wt[buf][(wave * 16 + lcol) * 32 + khB]);
    acc0 = wmma_bf16(a0, bb, acc0);
    acc1 = wmma_bf16(a1, bb, acc1);

    if (more) {
      __syncthreads();                  // all waves done reading wt[buf^1]
      STORE_TILE(buf ^ 1);
      __syncthreads();                  // stores visible before next compute
    }
  }
#undef LOAD_TILE
#undef STORE_TILE

  // Epilogue: add bias, store bf16 into transposed layouts.
  // j0 is block-uniform and FCHALF_ % 128 == 0 -> scalar branch, no exec mask.
  const int col  = j0 + wave * 16 + lcol;
  const float bi = bias[col];
  if (j0 < FCHALF_) {
    const int d = col >> 9, f = col & (HXR_ - 1);
    bf16_t* p = fc1t + (size_t)f * HX_ + d;
#pragma unroll
    for (int v = 0; v < 8; ++v) {
      const int m0 = v + ((lane < 16) ? 0 : 8);    // sample (D-layout row)
      p[(size_t)m0 * FCHALF_]        = (bf16_t)(acc0[v] + bi);
      p[(size_t)(m0 + 16) * FCHALF_] = (bf16_t)(acc1[v] + bi);
    }
  } else {
    const int jj = col - FCHALF_;
    const int f = jj >> 7, d = jj & (HX_ - 1);
    bf16_t* p = fc2t + (size_t)d * HXR_ + f;
#pragma unroll
    for (int v = 0; v < 8; ++v) {
      const int m0 = v + ((lane < 16) ? 0 : 8);
      p[(size_t)m0 * FCHALF_]        = (bf16_t)(acc0[v] + bi);
      p[(size_t)(m0 + 16) * FCHALF_] = (bf16_t)(acc1[v] + bi);
    }
  }
}

// ---------------------------------------------------------------------------
// Kernel 2: per-column inverse L2 norms (normalization == column scaling,
// so it commutes with the GEMMs and is applied as an epilogue scale).
// blocks [0, B*HXR)        -> invn1 (fc1 columns, reduce over d: 128 elems)
// blocks [B*HXR, +B*HX)    -> invn2 (fc2 columns, reduce over f: 512 elems)
// ---------------------------------------------------------------------------
__global__ __launch_bounds__(128) void col_norms(const bf16_t* __restrict__ fc1t,
                                                 const bf16_t* __restrict__ fc2t,
                                                 float* __restrict__ invn1,
                                                 float* __restrict__ invn2) {
  __shared__ float red[128];
  const int tid = threadIdx.x;
  const int row = blockIdx.x;
  float ss = 0.f;
  if (row < B_ * HXR_) {
    const float v = (float)fc1t[(size_t)row * HX_ + tid];
    ss = v * v;
  } else {
    const bf16_t* p = fc2t + (size_t)(row - B_ * HXR_) * HXR_;
#pragma unroll
    for (int i = 0; i < 4; ++i) {
      const float v = (float)p[tid + 128 * i];
      ss += v * v;
    }
  }
  red[tid] = ss;
  __syncthreads();
#pragma unroll
  for (int off = 64; off > 0; off >>= 1) {
    if (tid < off) red[tid] += red[tid + off];
    __syncthreads();
  }
  if (tid == 0) {
    const float inv = 1.0f / fmaxf(sqrtf(red[0]), 1e-12f);
    if (row < B_ * HXR_) invn1[row] = inv;
    else invn2[row - B_ * HXR_] = inv;
  }
}

// ---------------------------------------------------------------------------
// Kernel 3: fused RMSNorm -> GEMM1 -> scale -> silu -> GEMM2 -> scale -> +x
// Grid: (N/64, B). 8 waves; wave pair (2 per 16-row n-subtile) splits f / d.
// LDS: xs 16KB + ys 64KB = 80KB -> 2 workgroups per WGP.
// A-fragments are hoisted/reused: phase 2 loads its 4 A tiles once for all 16
// f-tiles; phase 3 holds 4 accumulators so each ys A tile is loaded once.
// ---------------------------------------------------------------------------
__global__ __launch_bounds__(256) void mlp_fused(const float* __restrict__ x,
                                                 const float* __restrict__ g,
                                                 const bf16_t* __restrict__ fc1t,
                                                 const bf16_t* __restrict__ fc2t,
                                                 const float* __restrict__ invn1,
                                                 const float* __restrict__ invn2,
                                                 float* __restrict__ out) {
  constexpr int MT = 64;
  __shared__ bf16_t xs[MT * HX_];    // rmsnormed x tile, [n][k]
  __shared__ bf16_t ys[MT * HXR_];   // silu(h@fc1) tile, [n][f]

  const int tid  = threadIdx.x;
  const int wave = tid >> 5;
  const int lane = tid & 31;
  const int b    = blockIdx.y;
  const int n0   = blockIdx.x * MT;
  const int lcol = lane & 15;
  const int khB  = (lane < 16) ? 0 : 16;

  const float* xb = x + ((size_t)b * N_ + n0) * HX_;

  // ---- Phase 1: RMSNorm rows -> bf16 LDS -------------------------------
  for (int r = wave; r < MT; r += 8) {
    const v4f xv = *(const v4f*)(xb + (size_t)r * HX_ + lane * 4);
    float ssq = xv.x * xv.x + xv.y * xv.y + xv.z * xv.z + xv.w * xv.w;
#pragma unroll
    for (int m = 16; m > 0; m >>= 1) ssq += __shfl_xor(ssq, m, 32);
    const float scale = rsqrtf(ssq * (1.0f / HX_) + 1e-6f);
    const v4f gv = *(const v4f*)(g + lane * 4);
    bf16_t* dst = xs + r * HX_ + lane * 4;
    dst[0] = (bf16_t)(xv.x * scale * gv.x);
    dst[1] = (bf16_t)(xv.y * scale * gv.y);
    dst[2] = (bf16_t)(xv.z * scale * gv.z);
    dst[3] = (bf16_t)(xv.w * scale * gv.w);
  }
  __syncthreads();

  const int nsub = wave >> 1;              // 16-row n-subtile (0..3)
  const int row0 = nsub * 16;

  // ---- Phase 2: y = rms(x) @ fc1, scale by invn1, silu, -> LDS ----------
  {
    const int fbase = (wave & 1) * 256;     // f half handled by this wave
    const bf16_t* fc1b = fc1t + (size_t)b * FCHALF_;

    v16bf aK[4];                            // whole K=128 of A, loaded once
#pragma unroll
    for (int k = 0; k < 4; ++k) aK[k] = load_a16x32(xs, HX_, row0, k * 32, lane);

    for (int ft = 0; ft < 16; ++ft) {
      const int f0 = fbase + ft * 16;
      const bf16_t* bp = fc1b + (size_t)(f0 + lcol) * HX_ + khB;
      v8f acc = {};
#pragma unroll
      for (int k = 0; k < 4; ++k)
        acc = wmma_bf16(aK[k], load_b32x16(bp + k * 32), acc);

      const int col = f0 + lcol;
      const float inv = invn1[b * HXR_ + col];
#pragma unroll
      for (int v = 0; v < 8; ++v) {
        const float yv = acc[v] * inv;
        const float sv = yv / (1.0f + __expf(-yv));     // silu
        const int row = row0 + v + ((lane < 16) ? 0 : 8);
        ys[row * HXR_ + col] = (bf16_t)sv;
      }
    }
  }
  __syncthreads();

  // ---- Phase 3: z = y @ fc2, scale by invn2, residual add, store --------
  {
    const int dbase = (wave & 1) * 64;      // d half handled by this wave
    const bf16_t* fc2b = fc2t + (size_t)b * FCHALF_;
    float* outb = out + ((size_t)b * N_ + n0) * HX_;

    v8f acc[4] = {};                        // 4 d-tiles accumulated at once
    for (int k0 = 0; k0 < HXR_; k0 += 32) {
      const v16bf a = load_a16x32(ys, HXR_, row0, k0, lane);   // one A read
#pragma unroll
      for (int dt = 0; dt < 4; ++dt) {
        const bf16_t* bp =
            fc2b + (size_t)(dbase + dt * 16 + lcol) * HXR_ + k0 + khB;
        acc[dt] = wmma_bf16(a, load_b32x16(bp), acc[dt]);
      }
    }
#pragma unroll
    for (int dt = 0; dt < 4; ++dt) {
      const int col = dbase + dt * 16 + lcol;
      const float inv = invn2[b * HX_ + col];
#pragma unroll
      for (int v = 0; v < 8; ++v) {
        const int row = row0 + v + ((lane < 16) ? 0 : 8);
        const float res = acc[dt][v] * inv + xb[(size_t)row * HX_ + col];
        __builtin_nontemporal_store(res, outb + (size_t)row * HX_ + col);
      }
    }
  }
}

// ---------------------------------------------------------------------------
// Launcher
// Inputs: x[32,4096,128] f32, s[32,1024] f32, W[1024,131072] f32,
//         bias[131072] f32, g[128] f32.  Output: [32,4096,128] f32.
// Workspace: fc1t 4MB + fc2t 4MB (bf16) + invn1 64KB + invn2 16KB.
// ---------------------------------------------------------------------------
extern "C" void kernel_launch(void* const* d_in, const int* in_sizes, int n_in,
                              void* d_out, int out_size, void* d_ws, size_t ws_size,
                              hipStream_t stream) {
  (void)in_sizes; (void)n_in; (void)out_size; (void)ws_size;
  const float* x    = (const float*)d_in[0];
  const float* s    = (const float*)d_in[1];
  const float* W    = (const float*)d_in[2];
  const float* bias = (const float*)d_in[3];
  const float* g    = (const float*)d_in[4];
  float* out = (float*)d_out;

  char* ws = (char*)d_ws;
  const size_t fcBytes = (size_t)B_ * FCHALF_ * sizeof(bf16_t);   // 4 MB each
  bf16_t* fc1t = (bf16_t*)ws;
  bf16_t* fc2t = (bf16_t*)(ws + fcBytes);
  float*  invn1 = (float*)(ws + 2 * fcBytes);
  float*  invn2 = invn1 + (size_t)B_ * HXR_;

  hyper_gemm<<<PCOLS_ / 128, 256, 0, stream>>>(s, W, bias, fc1t, fc2t);
  col_norms<<<B_ * HXR_ + B_ * HX_, 128, 0, stream>>>(fc1t, fc2t, invn1, invn2);
  dim3 grid(N_ / 64, B_);
  mlp_fused<<<grid, 256, 0, stream>>>(x, g, fc1t, fc2t, invn1, invn2, out);
}
